// DynamicQuantizeLinear_33174327394651
// MI455X (gfx1250) — compile-verified, matrix operand order checked by reading, not run on hardware
//
#include <hip/hip_runtime.h>
#include <hip/hip_bf16.h>
#include <stdint.h>

// ---- problem constants (fixed by the reference) ----
#define MDIM 8192
#define KDIM 4096
#define NDIM 11008
#define BM 128
#define BN 128
#define BK 32                  // = quant group size -> one scale row per K-step
#define NKB (KDIM / BK)        // 128 K-steps
#define LSTR (BK + 8)          // LDS row stride in bf16 (80B: 16B-aligned, bank-spread)

typedef __bf16 bf16_t;
typedef __attribute__((ext_vector_type(16))) __bf16 v16bf;
typedef __attribute__((ext_vector_type(8)))  float  v8f;
typedef __attribute__((ext_vector_type(4)))  float  f32x4;
typedef __attribute__((ext_vector_type(4)))  unsigned int u32x4;
typedef __attribute__((ext_vector_type(2)))  unsigned int u32x2;
typedef int v4i32 __attribute__((vector_size(16)));   // matches the async builtin's pointee

union Frag16 { v16bf v; u32x4 q[2]; };

// f32 pair -> packed bf16, RNE (used in the memory-bound prepass where cost is free)
__device__ __forceinline__ unsigned int pack2_bf16_rne(float a, float b) {
  unsigned int ua = __builtin_bit_cast(unsigned int, a);
  unsigned int ub = __builtin_bit_cast(unsigned int, b);
  ua += 0x7FFFu + ((ua >> 16) & 1u);
  ub += 0x7FFFu + ((ub >> 16) & 1u);
  return (ua >> 16) | (ub & 0xFFFF0000u);
}

// f32 pair -> packed bf16, round-nearest (ties up): 2x v_add + 1x v_perm_b32
__device__ __forceinline__ unsigned int pack2_bf16_fast(float a, float b) {
  unsigned int ua = __builtin_bit_cast(unsigned int, a) + 0x8000u;
  unsigned int ub = __builtin_bit_cast(unsigned int, b) + 0x8000u;
  // dst = { ub[3], ub[2], ua[3], ua[2] }  (a -> low half)
  return __builtin_amdgcn_perm(ub, ua, 0x07060302u);
}

// int4 magic: as_float(0x4B000000 | nib) == 2^23 + nib  (exact)
#define NIB_MAGIC 0x4B000000u
#define NIB_BIAS  8388616.0f   // 2^23 + 8 (exactly representable)

// ---- CDNA5 async copy global->LDS (ASYNCcnt) ----
__device__ __forceinline__ void async_copy_b128(const void* gsrc, unsigned int lds_off) {
#if __has_builtin(__builtin_amdgcn_global_load_async_to_lds_b128)
  typedef __attribute__((address_space(1))) v4i32 g_v4i32;
  typedef __attribute__((address_space(3))) v4i32 l_v4i32;
  __builtin_amdgcn_global_load_async_to_lds_b128(
      (g_v4i32*)(uintptr_t)gsrc, (l_v4i32*)(uintptr_t)lds_off, 0, 0);
#else
  unsigned long long ga = (unsigned long long)(uintptr_t)gsrc;
  asm volatile("global_load_async_to_lds_b128 %0, %1, off"
               :: "v"(lds_off), "v"(ga) : "memory");
#endif
}

__device__ __forceinline__ void wait_async0() {
#if __has_builtin(__builtin_amdgcn_s_wait_asynccnt)
  __builtin_amdgcn_s_wait_asynccnt(0);
#else
  asm volatile("s_wait_asynccnt 0" ::: "memory");
#endif
}

// =====================  prepass: A f32 -> bf16 (memory-bound)  ==========
__global__ __launch_bounds__(256)
void a_to_bf16(const float* __restrict__ A, bf16_t* __restrict__ Abf) {
  const size_t i = ((size_t)blockIdx.x * 256 + threadIdx.x) * 4;
  const f32x4 v = *(const f32x4*)(A + i);
  u32x2 t;
  t[0] = pack2_bf16_rne(v[0], v[1]);
  t[1] = pack2_bf16_rne(v[2], v[3]);
  *(u32x2*)(Abf + i) = t;
}

// =====================  main kernel: bf16-A + async LDS copy  =================
__global__ __launch_bounds__(256, 2)
void w4a32_wmma_gemm_async(const bf16_t* __restrict__ Abf,  // [M][K] bf16 (prepass)
                           const uint8_t* __restrict__ Wq,  // [K/2][N], lo nibble = even k
                           const float* __restrict__ Sc,    // [G][N]
                           const float* __restrict__ bias,  // [N]
                           float* __restrict__ out) {       // [M][N] f32
  __shared__ bf16_t shA[2][BM * LSTR];                      // [m][k] bf16
  __shared__ bf16_t shB[2][BN * LSTR];                      // [n][k] bf16 (transposed)

  const int tid  = threadIdx.x;
  const int lane = tid & 31;
  const int wave = tid >> 5;
  const int nb = blockIdx.x;
  const int mb = blockIdx.y;

  // ---- loader geometry ----
  const int rA = tid >> 2;         // A: 64 rows per pass, 2 passes (16B per thread-chunk)
  const int cA = (tid & 3) * 8;    // A: bf16 column {0,8,16,24}
  const int wi  = tid >> 4;        // W: packed-k row 0..15 within tile
  const int wn0 = (tid & 15) * 8;  // W: 8 bytes (8 n-columns) per thread

  const bf16_t*  Ag = Abf + (size_t)(mb * BM) * KDIM;
  const uint8_t* Wg = Wq  + (size_t)(nb * BN);
  const float*   Sg = Sc  + (size_t)(nb * BN);

  // ---- compute geometry: 8 waves in 4(M) x 2(N); each wave 2x4 16x16 tiles ----
  const int wm   = wave & 3;
  const int wn   = wave >> 2;
  const int half = lane >> 4;
  const int r    = lane & 15;
  const int ko   = half * 8;

  v8f acc[2][4];
  #pragma unroll
  for (int i = 0; i < 2; ++i)
    #pragma unroll
    for (int j = 0; j < 4; ++j)
      #pragma unroll
      for (int v = 0; v < 8; ++v) acc[i][j][v] = 0.0f;

  u32x2 w_reg;
  f32x4 s_reg[2];

  auto issue_asyncA = [&](int kb, int buf) {     // pure DMA: no VALU, no ds_store
    #pragma unroll
    for (int p = 0; p < 2; ++p) {
      const int row = p * 64 + rA;
      const bf16_t* src = Ag + (size_t)row * KDIM + (size_t)kb * BK + cA;
      const unsigned int dst = (unsigned int)(uintptr_t)&shA[buf][row * LSTR + cA];
      async_copy_b128(src, dst);
    }
  };

  auto stage_globalW = [&](int kb) {
    w_reg    = *(const u32x2*)(Wg + (size_t)(kb * (BK / 2) + wi) * NDIM + wn0);
    s_reg[0] = *(const f32x4*)(Sg + (size_t)kb * NDIM + wn0);      // group g == kb
    s_reg[1] = *(const f32x4*)(Sg + (size_t)kb * NDIM + wn0 + 4);
  };

  auto stage_ldsW = [&](int buf) {
    bf16_t* sB = shB[buf];
    #pragma unroll
    for (int j = 0; j < 8; ++j) {            // int4 dequant -> bf16, [n][k]
      const unsigned int w  = w_reg[j >> 2];
      const unsigned int sh = (j & 3) * 8;
      const float sc = (j < 4) ? s_reg[0][j & 3] : s_reg[1][j & 3];
      const float c  = -NIB_BIAS * sc;       // folds the (-8) zero-point
      const float lo = __builtin_fmaf(
          __builtin_bit_cast(float, NIB_MAGIC | ((w >> sh) & 15u)), sc, c);       // k = 2*wi
      const float hi = __builtin_fmaf(
          __builtin_bit_cast(float, NIB_MAGIC | ((w >> (sh + 4)) & 15u)), sc, c); // k = 2*wi+1
      *(unsigned int*)&sB[(wn0 + j) * LSTR + 2 * wi] = pack2_bf16_fast(lo, hi);
    }
  };

  auto compute = [&](int buf) {
    const bf16_t* sA = shA[buf];
    const bf16_t* sB = shB[buf];
    Frag16 af[2], bfr[4];
    // 16-bit A layout (ISA 7.12.2): lanes 0-15 get K 0-7|16-23, lanes 16-31 get K 8-15|24-31
    #pragma unroll
    for (int i = 0; i < 2; ++i) {
      const int row = (wm * 2 + i) * 16 + r;
      af[i].q[0] = *(const u32x4*)&sA[row * LSTR + ko];        // ds_load_b128
      af[i].q[1] = *(const u32x4*)&sA[row * LSTR + ko + 16];
    }
    #pragma unroll
    for (int j = 0; j < 4; ++j) {
      const int col = (wn * 4 + j) * 16 + r;
      bfr[j].q[0] = *(const u32x4*)&sB[col * LSTR + ko];
      bfr[j].q[1] = *(const u32x4*)&sB[col * LSTR + ko + 16];
    }
    #pragma unroll
    for (int i = 0; i < 2; ++i)
      #pragma unroll
      for (int j = 0; j < 4; ++j)
        acc[i][j] = __builtin_amdgcn_wmma_f32_16x16x32_bf16(
            false, af[i].v, false, bfr[j].v, (short)0, acc[i][j], false, false);
  };

  // ---- software-pipelined main loop ----
  issue_asyncA(0, 0);
  stage_globalW(0);
  stage_ldsW(0);
  wait_async0();
  __syncthreads();
  for (int kb = 0; kb < NKB; ++kb) {
    if (kb + 1 < NKB) {
      issue_asyncA(kb + 1, (kb + 1) & 1);   // DMA overlaps compute below
      stage_globalW(kb + 1);
    }
    compute(kb & 1);
    if (kb + 1 < NKB) stage_ldsW((kb + 1) & 1);
    wait_async0();                          // next A tile fully landed in LDS
    __syncthreads();
  }

  // ---- epilogue: C layout VGPR v -> row v + 8*half, col = lane&15 ----
  #pragma unroll
  for (int i = 0; i < 2; ++i) {
    const int gm0 = mb * BM + (wm * 2 + i) * 16 + half * 8;
    #pragma unroll
    for (int j = 0; j < 4; ++j) {
      const int gn = nb * BN + (wn * 4 + j) * 16 + r;
      const float bv = bias[gn];
      #pragma unroll
      for (int v = 0; v < 8; ++v)
        out[(size_t)(gm0 + v) * NDIM + gn] = acc[i][j][v] + bv;
    }
  }
}

// =====================  fallback (ws too small): fused conversion  ============
__global__ __launch_bounds__(256, 2)
void w4a32_wmma_gemm_fused(const float* __restrict__ A,
                           const uint8_t* __restrict__ Wq,
                           const float* __restrict__ Sc,
                           const float* __restrict__ bias,
                           float* __restrict__ out) {
  __shared__ bf16_t shA[2][BM * LSTR];
  __shared__ bf16_t shB[2][BN * LSTR];

  const int tid  = threadIdx.x;
  const int lane = tid & 31;
  const int wave = tid >> 5;
  const int nb = blockIdx.x;
  const int mb = blockIdx.y;

  const int rA  = tid >> 3;
  const int kA0 = (tid & 7) * 4;
  const int wi  = tid >> 4;
  const int wn0 = (tid & 15) * 8;

  const float*   Ag = A  + (size_t)(mb * BM) * KDIM;
  const uint8_t* Wg = Wq + (size_t)(nb * BN);
  const float*   Sg = Sc + (size_t)(nb * BN);

  const int wm   = wave & 3;
  const int wn   = wave >> 2;
  const int half = lane >> 4;
  const int r    = lane & 15;
  const int ko   = half * 8;

  v8f acc[2][4];
  #pragma unroll
  for (int i = 0; i < 2; ++i)
    #pragma unroll
    for (int j = 0; j < 4; ++j)
      #pragma unroll
      for (int v = 0; v < 8; ++v) acc[i][j][v] = 0.0f;

  f32x4 a_reg[4];
  u32x2 w_reg;
  f32x4 s_reg[2];

  auto stage_global = [&](int kb) {
    #pragma unroll
    for (int p = 0; p < 4; ++p)
      a_reg[p] = *(const f32x4*)(Ag + (size_t)(p * 32 + rA) * KDIM + (size_t)kb * BK + kA0);
    w_reg    = *(const u32x2*)(Wg + (size_t)(kb * (BK / 2) + wi) * NDIM + wn0);
    s_reg[0] = *(const f32x4*)(Sg + (size_t)kb * NDIM + wn0);
    s_reg[1] = *(const f32x4*)(Sg + (size_t)kb * NDIM + wn0 + 4);
  };

  auto stage_lds = [&](int buf) {
    bf16_t* sA = shA[buf];
    bf16_t* sB = shB[buf];
    #pragma unroll
    for (int p = 0; p < 4; ++p) {
      u32x2 t;
      t[0] = pack2_bf16_fast(a_reg[p][0], a_reg[p][1]);
      t[1] = pack2_bf16_fast(a_reg[p][2], a_reg[p][3]);
      *(u32x2*)&sA[(p * 32 + rA) * LSTR + kA0] = t;
    }
    #pragma unroll
    for (int j = 0; j < 8; ++j) {
      const unsigned int w  = w_reg[j >> 2];
      const unsigned int sh = (j & 3) * 8;
      const float sc = (j < 4) ? s_reg[0][j & 3] : s_reg[1][j & 3];
      const float c  = -NIB_BIAS * sc;
      const float lo = __builtin_fmaf(
          __builtin_bit_cast(float, NIB_MAGIC | ((w >> sh) & 15u)), sc, c);
      const float hi = __builtin_fmaf(
          __builtin_bit_cast(float, NIB_MAGIC | ((w >> (sh + 4)) & 15u)), sc, c);
      *(unsigned int*)&sB[(wn0 + j) * LSTR + 2 * wi] = pack2_bf16_fast(lo, hi);
    }
  };

  auto compute = [&](int buf) {
    const bf16_t* sA = shA[buf];
    const bf16_t* sB = shB[buf];
    Frag16 af[2], bfr[4];
    #pragma unroll
    for (int i = 0; i < 2; ++i) {
      const int row = (wm * 2 + i) * 16 + r;
      af[i].q[0] = *(const u32x4*)&sA[row * LSTR + ko];
      af[i].q[1] = *(const u32x4*)&sA[row * LSTR + ko + 16];
    }
    #pragma unroll
    for (int j = 0; j < 4; ++j) {
      const int col = (wn * 4 + j) * 16 + r;
      bfr[j].q[0] = *(const u32x4*)&sB[col * LSTR + ko];
      bfr[j].q[1] = *(const u32x4*)&sB[col * LSTR + ko + 16];
    }
    #pragma unroll
    for (int i = 0; i < 2; ++i)
      #pragma unroll
      for (int j = 0; j < 4; ++j)
        acc[i][j] = __builtin_amdgcn_wmma_f32_16x16x32_bf16(
            false, af[i].v, false, bfr[j].v, (short)0, acc[i][j], false, false);
  };

  stage_global(0);
  stage_lds(0);
  __syncthreads();
  for (int kb = 0; kb < NKB; ++kb) {
    if (kb + 1 < NKB) stage_global(kb + 1);
    compute(kb & 1);
    if (kb + 1 < NKB) stage_lds((kb + 1) & 1);
    __syncthreads();
  }

  #pragma unroll
  for (int i = 0; i < 2; ++i) {
    const int gm0 = mb * BM + (wm * 2 + i) * 16 + half * 8;
    #pragma unroll
    for (int j = 0; j < 4; ++j) {
      const int gn = nb * BN + (wn * 4 + j) * 16 + r;
      const float bv = bias[gn];
      #pragma unroll
      for (int v = 0; v < 8; ++v)
        out[(size_t)(gm0 + v) * NDIM + gn] = acc[i][j][v] + bv;
    }
  }
}

extern "C" void kernel_launch(void* const* d_in, const int* in_sizes, int n_in,
                              void* d_out, int out_size, void* d_ws, size_t ws_size,
                              hipStream_t stream) {
  (void)in_sizes; (void)n_in; (void)out_size;
  const float*   A    = (const float*)d_in[0];
  const uint8_t* Wq   = (const uint8_t*)d_in[1];
  const float*   Sc   = (const float*)d_in[2];
  const float*   bias = (const float*)d_in[3];
  float* out = (float*)d_out;

  dim3 grid(NDIM / BN, MDIM / BM);   // 86 x 64
  dim3 block(256);                   // 8 wave32

  const size_t needA = (size_t)MDIM * KDIM * sizeof(bf16_t);   // 64 MiB
  if (ws_size >= needA) {
    bf16_t* Abf = (bf16_t*)d_ws;
    a_to_bf16<<<(MDIM * (size_t)KDIM) / (4 * 256), 256, 0, stream>>>(A, Abf);
    w4a32_wmma_gemm_async<<<grid, block, 0, stream>>>(Abf, Wq, Sc, bias, out);
  } else {
    w4a32_wmma_gemm_fused<<<grid, block, 0, stream>>>(A, Wq, Sc, bias, out);
  }
}
